// SigLipLossWithAmbiguityForOpenClip_23021024706697
// MI455X (gfx1250) — compile-verified, matrix operand order checked by reading, not run on hardware
//
#include <hip/hip_runtime.h>
#include <hip/hip_bf16.h>
#include <math.h>

typedef __attribute__((ext_vector_type(16))) _Float16 v16h;
typedef __attribute__((ext_vector_type(8)))  _Float16 v8h;
typedef __attribute__((ext_vector_type(8)))  float    v8f;

#define S_IMG 65536
#define N_TXT 8192
#define D_DIM 128
#define TILES_PER_DIM (N_TXT / 16)              // 512
#define N_TILES (TILES_PER_DIM * TILES_PER_DIM) // 262144
#define WMMA_BLOCKS (N_TILES / 8)               // 32768

// ---- workspace layout (bytes) ----
#define OFF_ZTXT   ((size_t)0)                                   // N*D f16
#define OFF_ZSEL   (OFF_ZTXT + (size_t)N_TXT * D_DIM * 2)        // N*D f16
#define OFF_POT    (OFF_ZSEL + (size_t)N_TXT * D_DIM * 2)        // S f32
#define OFF_SEGMIN (OFF_POT + (size_t)S_IMG * 4)                 // N u32
#define OFF_BEST   (OFF_SEGMIN + (size_t)N_TXT * 4)              // N u32
#define OFF_VALID  (OFF_BEST + (size_t)N_TXT * 4)                // N i32
#define OFF_NV     (OFF_VALID + (size_t)N_TXT * 4)               // 1 i32 (padded 16B)
#define OFF_PART   (OFF_NV + 16)                                 // WMMA_BLOCKS f32

__device__ __forceinline__ float wave_sum(float v) {
#pragma unroll
    for (int off = 16; off > 0; off >>= 1) v += __shfl_xor(v, off, 32);
    return v;
}

__device__ __forceinline__ float softplus_f(float x) {
    return fmaxf(x, 0.0f) + log1pf(expf(-fabsf(x)));
}

// ---------- init scratch state ----------
__global__ void k_init(unsigned* seg_min, unsigned* best, int* valid, int* num_valid) {
    int i = blockIdx.x * blockDim.x + threadIdx.x;
    if (i < N_TXT) { seg_min[i] = 0x7F800000u; best[i] = S_IMG; valid[i] = 0; }
    if (i == 0) *num_valid = 0;
}

// ---------- normalize text features -> f16 (one wave per row) ----------
__global__ void k_norm_text(const float* __restrict__ txt, _Float16* __restrict__ ztxt) {
    const int wave = threadIdx.x >> 5, lane = threadIdx.x & 31;
    const int row = blockIdx.x * 8 + wave;
    const float4 v = *(const float4*)(txt + (size_t)row * D_DIM + lane * 4);
    float ss = wave_sum(v.x * v.x + v.y * v.y + v.z * v.z + v.w * v.w);
    const float inv = 1.0f / (sqrtf(ss) + 1e-12f);
    _Float16* o = ztxt + (size_t)row * D_DIM + lane * 4;
    o[0] = (_Float16)(v.x * inv); o[1] = (_Float16)(v.y * inv);
    o[2] = (_Float16)(v.z * inv); o[3] = (_Float16)(v.w * inv);
}

// ---------- per-image true-positive loss + segmented min (one wave per image) ----------
__global__ void k_image_pot(const float* __restrict__ img, const _Float16* __restrict__ ztxt,
                            const int* __restrict__ key,
                            const float* __restrict__ scale_p, const float* __restrict__ bias_p,
                            float* __restrict__ pot, unsigned* __restrict__ seg_min) {
    const int wave = threadIdx.x >> 5, lane = threadIdx.x & 31;
    const int s = blockIdx.x * 8 + wave;
    const int k = key[s];
    const float4 v = *(const float4*)(img + (size_t)s * D_DIM + lane * 4);
    float ss = wave_sum(v.x * v.x + v.y * v.y + v.z * v.z + v.w * v.w);
    const float inv = 1.0f / (sqrtf(ss) + 1e-12f);
    const _Float16* t = ztxt + (size_t)k * D_DIM + lane * 4;
    float d = v.x * (float)t[0] + v.y * (float)t[1] + v.z * (float)t[2] + v.w * (float)t[3];
    d = wave_sum(d);
    const float tp = d * inv * (*scale_p) + (*bias_p);
    const float p = softplus_f(-tp);
    if (lane == 0) {
        pot[s] = p;
        atomicMin(&seg_min[k], __float_as_uint(p));  // p >= 0 -> uint order == float order
    }
}

// ---------- first-occurrence argmin per segment ----------
__global__ void k_pick_best(const float* __restrict__ pot, const int* __restrict__ key,
                            const unsigned* __restrict__ seg_min, unsigned* __restrict__ best) {
    const int s = blockIdx.x * blockDim.x + threadIdx.x;
    if (s >= S_IMG) return;
    const int k = key[s];
    if (__float_as_uint(pot[s]) == seg_min[k]) atomicMin(&best[k], (unsigned)s);
}

// ---------- gather selected images, normalize -> f16 (one wave per text) ----------
__global__ void k_gather(const float* __restrict__ img, const unsigned* __restrict__ seg_min,
                         const unsigned* __restrict__ best, _Float16* __restrict__ zsel,
                         int* __restrict__ valid, int* __restrict__ num_valid) {
    const int wave = threadIdx.x >> 5, lane = threadIdx.x & 31;
    const int n = blockIdx.x * 8 + wave;
    _Float16* o = zsel + (size_t)n * D_DIM + lane * 4;
    if (seg_min[n] != 0x7F800000u) {   // uniform per wave
        const unsigned b = best[n];
        const float4 v = *(const float4*)(img + (size_t)b * D_DIM + lane * 4);
        float ss = wave_sum(v.x * v.x + v.y * v.y + v.z * v.z + v.w * v.w);
        const float inv = 1.0f / (sqrtf(ss) + 1e-12f);
        o[0] = (_Float16)(v.x * inv); o[1] = (_Float16)(v.y * inv);
        o[2] = (_Float16)(v.z * inv); o[3] = (_Float16)(v.w * inv);
        if (lane == 0) { valid[n] = 1; atomicAdd(num_valid, 1); }
    } else {
        o[0] = (_Float16)0.f; o[1] = (_Float16)0.f; o[2] = (_Float16)0.f; o[3] = (_Float16)0.f;
        if (lane == 0) valid[n] = 0;
    }
}

// ---------- WMMA n x n logits + softplus epilogue (one 16x16 tile per wave) ----------
__global__ void k_wmma_loss(const _Float16* __restrict__ zsel, const _Float16* __restrict__ ztxt,
                            const int* __restrict__ valid,
                            const float* __restrict__ scale_p, const float* __restrict__ bias_p,
                            float* __restrict__ partials) {
    const int wave = threadIdx.x >> 5;
    const int lane = threadIdx.x & 31;
    const int tile = blockIdx.x * 8 + wave;
    const int ti = tile >> 9;            // tile / 512
    const int tj = tile & 511;
    const int i0 = ti * 16, j0 = tj * 16;
    const int m = lane & 15;             // A row within tile; also C/D column index n
    const int khalf = lane >> 4;

    const _Float16* arow = zsel + (size_t)(i0 + m) * D_DIM;
    const _Float16* brow = ztxt + (size_t)(j0 + m) * D_DIM;  // column j0+n of B = row of ztxt

    // Two independent accumulator chains to break the WMMA->WMMA RAW hazard.
    v8f cA = {}, cB = {};
#pragma unroll
    for (int cc = 0; cc < 4; ++cc) {
        const int kb = cc * 32 + khalf * 8;
        const v8h alo = *(const v8h*)(arow + kb);
        const v8h ahi = *(const v8h*)(arow + kb + 16);
        const v8h blo = *(const v8h*)(brow + kb);
        const v8h bhi = *(const v8h*)(brow + kb + 16);
        v16h a, b;
#pragma unroll
        for (int i = 0; i < 8; ++i) {
            a[i] = alo[i]; a[i + 8] = ahi[i];
            b[i] = blo[i]; b[i + 8] = bhi[i];
        }
        if (cc & 1)
            cB = __builtin_amdgcn_wmma_f32_16x16x32_f16(false, a, false, b, (short)0, cB,
                                                        false, false);
        else
            cA = __builtin_amdgcn_wmma_f32_16x16x32_f16(false, a, false, b, (short)0, cA,
                                                        false, false);
    }
    v8f c;
#pragma unroll
    for (int i = 0; i < 8; ++i) c[i] = cA[i] + cB[i];

    const float scale = *scale_p, bias = *bias_p;
    const int gj = j0 + m;
    const float vj = (float)valid[gj];
    const int ibase = i0 + khalf * 8;
    float acc = 0.0f;
#pragma unroll
    for (int r = 0; r < 8; ++r) {
        const int gi = ibase + r;
        const float vi = (float)valid[gi];         // branchless mask
        const float logit = fmaf(c[r], scale, bias);
        const float x = (gi == gj) ? -logit : logit;
        acc += vj * vi * softplus_f(x);
    }
    acc = wave_sum(acc);

    __shared__ float smem[8];
    if (lane == 0) smem[wave] = acc;
    __syncthreads();
    if (threadIdx.x == 0) {
        float s = 0.0f;
#pragma unroll
        for (int w = 0; w < 8; ++w) s += smem[w];
        partials[blockIdx.x] = s;
    }
}

// ---------- deterministic final reduction ----------
__global__ void k_reduce(const float* __restrict__ partials, const int* __restrict__ num_valid,
                         float* __restrict__ out) {
    __shared__ float smem[256];
    float s = 0.0f;
    for (int i = threadIdx.x; i < WMMA_BLOCKS; i += 256) s += partials[i];
    smem[threadIdx.x] = s;
    __syncthreads();
    for (int off = 128; off > 0; off >>= 1) {
        if ((int)threadIdx.x < off) smem[threadIdx.x] += smem[threadIdx.x + off];
        __syncthreads();
    }
    if (threadIdx.x == 0) {
        int nv = *num_valid;
        if (nv < 1) nv = 1;
        out[0] = smem[0] / (float)nv;
    }
}

extern "C" void kernel_launch(void* const* d_in, const int* in_sizes, int n_in,
                              void* d_out, int out_size, void* d_ws, size_t ws_size,
                              hipStream_t stream) {
    const float* img     = (const float*)d_in[0];
    const float* txt     = (const float*)d_in[1];
    const int*   key     = (const int*)d_in[2];
    const float* scale_p = (const float*)d_in[3];
    const float* bias_p  = (const float*)d_in[4];
    float* out = (float*)d_out;

    char* ws = (char*)d_ws;
    _Float16* ztxt     = (_Float16*)(ws + OFF_ZTXT);
    _Float16* zsel     = (_Float16*)(ws + OFF_ZSEL);
    float*    pot      = (float*)   (ws + OFF_POT);
    unsigned* seg_min  = (unsigned*)(ws + OFF_SEGMIN);
    unsigned* best     = (unsigned*)(ws + OFF_BEST);
    int*      valid    = (int*)     (ws + OFF_VALID);
    int*      num_val  = (int*)     (ws + OFF_NV);
    float*    partials = (float*)   (ws + OFF_PART);

    k_init<<<(N_TXT + 255) / 256, 256, 0, stream>>>(seg_min, best, valid, num_val);
    k_norm_text<<<N_TXT / 8, 256, 0, stream>>>(txt, ztxt);
    k_image_pot<<<S_IMG / 8, 256, 0, stream>>>(img, ztxt, key, scale_p, bias_p, pot, seg_min);
    k_pick_best<<<S_IMG / 256, 256, 0, stream>>>(pot, key, seg_min, best);
    k_gather<<<N_TXT / 8, 256, 0, stream>>>(img, seg_min, best, zsel, valid, num_val);
    k_wmma_loss<<<WMMA_BLOCKS, 256, 0, stream>>>(zsel, ztxt, valid, scale_p, bias_p, partials);
    k_reduce<<<1, 256, 0, stream>>>(partials, num_val, out);
}
